// TrajGen_31988916421070
// MI455X (gfx1250) — matrix operand order repaired and checked by reading, not compiled, not run on hardware
//
#include <hip/hip_runtime.h>
#include <hip/hip_bf16.h>
#include <stdint.h>

// ---------------------------------------------------------------------------
// Problem constants
// ---------------------------------------------------------------------------
#define BATCH   512
#define NDOF    8
#define TLEN    4096
#define NCOND   4
#define H1      64
#define H2      128
#define NOUT    (NDOF * TLEN)      // 32768
#define NROWS   (BATCH * NDOF)     // 4096
#define NCH     16                 // scan chunks
#define TC      (TLEN / NCH)       // 256 steps per chunk

// gfx1250 async global<->LDS path (ASYNCcnt) via inline asm
#define USE_ASYNC_LDS 1

typedef __attribute__((ext_vector_type(16))) _Float16 v16h;
typedef __attribute__((ext_vector_type(8)))  _Float16 v8h;
typedef __attribute__((ext_vector_type(8)))  float    v8f;

// ---------------------------------------------------------------------------
// Kernel 1: h1 = relu([z | emb[cls]] @ W1 + b1)    (512 x 64)
// ---------------------------------------------------------------------------
__global__ void __launch_bounds__(256)
k_mlp1(const int* __restrict__ cls, const float* __restrict__ z,
       const float* __restrict__ emb, const float* __restrict__ W1,
       const float* __restrict__ b1, float* __restrict__ h1)
{
    int idx = blockIdx.x * blockDim.x + threadIdx.x;   // 512*64
    int b = idx >> 6, j = idx & 63;
    const float* zb = z + b * 64;
    float acc = b1[j];
#pragma unroll 8
    for (int k = 0; k < 64; ++k)
        acc = fmaf(zb[k], W1[k * 64 + j], acc);
    const float* y = emb + cls[b] * NCOND;
#pragma unroll
    for (int k = 0; k < NCOND; ++k)
        acc = fmaf(y[k], W1[(64 + k) * 64 + j], acc);
    h1[idx] = fmaxf(acc, 0.0f);
}

// ---------------------------------------------------------------------------
// Kernel 2: h2 = relu(h1 @ W2 + b2) -> f16   (512 x 128)
// ---------------------------------------------------------------------------
__global__ void __launch_bounds__(256)
k_mlp2(const float* __restrict__ h1, const float* __restrict__ W2,
       const float* __restrict__ b2, _Float16* __restrict__ h2h)
{
    int idx = blockIdx.x * blockDim.x + threadIdx.x;   // 512*128
    int b = idx >> 7, j = idx & 127;
    const float* hb = h1 + b * 64;
    float acc = b2[j];
#pragma unroll 8
    for (int k = 0; k < 64; ++k)
        acc = fmaf(hb[k], W2[k * 128 + j], acc);
    h2h[idx] = (_Float16)fmaxf(acc, 0.0f);
}

// ---------------------------------------------------------------------------
// Kernel 3: pack W3 (f32, 128 x 32768 row-major) into f16 WMMA-B fragment
// order. For N-tile tn (16 cols) and K-chunk c (32 rows), lane l holds
// column n = tn*16 + (l&15), rows k = c*32 + (l>>4)*16 .. +15, K-ascending,
// stored as 16 contiguous halves: W3p[((tn*4 + c)*32 + l)*16 + e].
// ---------------------------------------------------------------------------
__global__ void __launch_bounds__(256)
k_packW3(const float* __restrict__ W3, _Float16* __restrict__ W3p)
{
    int idx  = blockIdx.x * blockDim.x + threadIdx.x;  // 2048*4*32
    int lane = idx & 31;
    int c    = (idx >> 5) & 3;
    int tn   = idx >> 7;                               // 0..2047
    int n    = tn * 16 + (lane & 15);
    int kb   = c * 32 + (lane >> 4) * 16;
    _Float16 buf[16];
#pragma unroll
    for (int e = 0; e < 16; ++e)
        buf[e] = (_Float16)W3[(size_t)(kb + e) * NOUT + n];
    _Float16* dst = W3p + ((size_t)(tn * 4 + c) * 32 + lane) * 16;
    *(v8h*)(dst)     = *(v8h*)(&buf[0]);
    *(v8h*)(dst + 8) = *(v8h*)(&buf[8]);
}

// ---------------------------------------------------------------------------
// Kernel 4: GEMM3 via WMMA.  torque[b][n] = sigmoid(h2[b]·W3[:,n] + b3[n])
//                                           * (mx[d]-mn[d]) + mn[d],  d=n>>12
// One wave computes a 16(M) x 64(N) output strip: 4 accumulators, K=128 as
// 4 chunks of 32 -> 16 v_wmma_f32_16x16x32_f16 per wave job.
// ---------------------------------------------------------------------------
__global__ void __launch_bounds__(256)
k_gemm3(const _Float16* __restrict__ Ah, const _Float16* __restrict__ Bp,
        const float* __restrict__ b3, const float* __restrict__ mn,
        const float* __restrict__ mx, float* __restrict__ torque)
{
    int wave = (blockIdx.x * blockDim.x + threadIdx.x) >> 5;  // 16384 jobs
    int lane = threadIdx.x & 31;
    int mt = wave >> 9;            // 0..31
    int ng = wave & 511;           // 0..511 (groups of 4 N-tiles)
    int m0 = mt << 4, n0 = ng << 6;
    int nlo = lane & 15, hi = lane >> 4;

    // A: row (m0+nlo), elements at K offset hi*16 .. +15 (contiguous)
    const _Float16* arow = Ah + (size_t)(m0 + nlo) * H2 + hi * 16;

    v8f acc[4] = {{}, {}, {}, {}};

#pragma unroll
    for (int c = 0; c < 4; ++c) {
        const _Float16* ap = arow + c * 32;
        v8h alo = *(const v8h*)(ap);
        v8h ahi = *(const v8h*)(ap + 8);
        v16h afrag = __builtin_shufflevector(alo, ahi,
            0, 1, 2, 3, 4, 5, 6, 7, 8, 9, 10, 11, 12, 13, 14, 15);

        const _Float16* bbase =
            Bp + ((size_t)(ng * 16 + c) * 32 + lane) * 16;   // j stride: 4*512 halves
        if (c < 3)
            __builtin_prefetch(bbase + 512, 0, 0);           // next K-chunk

#pragma unroll
        for (int j = 0; j < 4; ++j) {
            const _Float16* bp = bbase + (size_t)j * 2048;
            v8h blo = *(const v8h*)(bp);
            v8h bhi = *(const v8h*)(bp + 8);
            v16h bfrag = __builtin_shufflevector(blo, bhi,
                0, 1, 2, 3, 4, 5, 6, 7, 8, 9, 10, 11, 12, 13, 14, 15);
            acc[j] = __builtin_amdgcn_wmma_f32_16x16x32_f16(
                false, afrag, false, bfrag, (short)0, acc[j], false, false);
        }
    }

    // Epilogue: bias + sigmoid + denormalize, coalesced row stores.
#pragma unroll
    for (int j = 0; j < 4; ++j) {
        int n   = n0 + j * 16 + nlo;
        int dof = n >> 12;                      // tile never crosses a dof
        float lo = mn[dof];
        float sc = mx[dof] - lo;
        float bias = b3[n];
        float* orow = torque + (size_t)(m0 + hi * 8) * NOUT + n;
#pragma unroll
        for (int r = 0; r < 8; ++r) {
            float v = acc[j][r] + bias;
            float s = 1.0f / (1.0f + __expf(-v));
            orow[(size_t)r * NOUT] = fmaf(s, sc, lo);
        }
    }
}

// ---------------------------------------------------------------------------
// Async staging helpers (gfx1250 GLOBAL_*_ASYNC_*_LDS via inline asm).
// vdst = per-lane LDS byte address, vaddr = 32-bit byte offset,
// saddr = uniform 64-bit base.  Tracked by ASYNCcnt.
// ---------------------------------------------------------------------------
__device__ __forceinline__ void async_g2l_b32(uint32_t lds_addr, uint32_t goff,
                                              const void* base)
{
    asm volatile("global_load_async_to_lds_b32 %0, %1, %2"
                 :: "v"(lds_addr), "v"(goff), "s"(base) : "memory");
}
__device__ __forceinline__ void async_l2g_b32(uint32_t goff, uint32_t lds_addr,
                                              void* base)
{
    asm volatile("global_store_async_from_lds_b32 %0, %1, %2"
                 :: "v"(goff), "v"(lds_addr), "s"(base) : "memory");
}
__device__ __forceinline__ void wait_async0()
{
    asm volatile("s_wait_asynccnt 0x0" ::: "memory");
}

// ---------------------------------------------------------------------------
// Scan pass 1: run each 256-step chunk from zero state -> affine offset q_c.
// job = row*NCH + chunk; torque addr base = job*TC (layouts line up exactly).
// Tiles staged into LDS with async global->LDS (coalesced 128B per wave op).
// ---------------------------------------------------------------------------
__global__ void __launch_bounds__(256)
k_scan1(const float* __restrict__ torque, const float* __restrict__ goal,
        float* __restrict__ qx, float* __restrict__ qdx)
{
    __shared__ float lds[256 * 33];
    int tid  = threadIdx.x;
    int lane = tid & 31, wv = tid >> 5;
    int job  = blockIdx.x * 256 + tid;
    int dof  = (job >> 4) & 7;
    float g  = goal[dof];
    const float dt = 0.01f, ax = 25.0f, bx = 6.25f;
    float x = 0.0f, dx = 0.0f;

    uint32_t ldsBase   = (uint32_t)(uintptr_t)(&lds[0]);
    uint32_t blockByte = (uint32_t)(blockIdx.x * 256 * TC) * 4u;

    for (int w = 0; w < TC / 32; ++w) {                    // 8 windows of 32
#if USE_ASYNC_LDS
#pragma unroll 8
        for (int i = 0; i < 32; ++i) {
            int jl = wv * 32 + i;
            async_g2l_b32(ldsBase + (uint32_t)(jl * 33 + lane) * 4u,
                          blockByte + (uint32_t)(jl * TC + w * 32 + lane) * 4u,
                          torque);
        }
        wait_async0();
#else
#pragma unroll 4
        for (int i = 0; i < 32; ++i) {
            int jl = wv * 32 + i;
            lds[jl * 33 + lane] =
                torque[(size_t)blockIdx.x * 256 * TC + (size_t)jl * TC + w * 32 + lane];
        }
#endif
        __syncthreads();
        const float* fr = &lds[tid * 33];
#pragma unroll
        for (int tt = 0; tt < 32; ++tt) {
            float ddx = ax * (bx * (g - x) - dx) + fr[tt];
            dx += ddx * dt;
            x  += dx * dt;
        }
        __syncthreads();
    }
    qx[job]  = x;
    qdx[job] = dx;
}

// ---------------------------------------------------------------------------
// Scan pass 2: per-row serial prefix over 16 chunks using M^256 (host-computed
// in double).  Writes each chunk's true start state.
// ---------------------------------------------------------------------------
__global__ void __launch_bounds__(256)
k_scan2(const float* __restrict__ qx, const float* __restrict__ qdx,
        const float* __restrict__ x0, float* __restrict__ sx,
        float* __restrict__ sdx, float m00, float m01, float m10, float m11)
{
    int row = blockIdx.x * blockDim.x + threadIdx.x;       // 4096
    int dof = row & 7;
    float x = x0[dof], dx = 0.0f;
#pragma unroll
    for (int c = 0; c < NCH; ++c) {
        int j = row * NCH + c;
        sx[j]  = x;
        sdx[j] = dx;
        float nx  = m00 * x + m01 * dx + qx[j];
        float ndx = m10 * x + m11 * dx + qdx[j];
        x = nx; dx = ndx;
    }
}

// ---------------------------------------------------------------------------
// Scan pass 3: replay each chunk from its corrected start, writing x.
// Loads async global->LDS; results written back via async LDS->global.
// ---------------------------------------------------------------------------
__global__ void __launch_bounds__(256)
k_scan3(const float* __restrict__ torque, const float* __restrict__ goal,
        const float* __restrict__ sx, const float* __restrict__ sdx,
        float* __restrict__ out)
{
    __shared__ float lds[256 * 33];
    int tid  = threadIdx.x;
    int lane = tid & 31, wv = tid >> 5;
    int job  = blockIdx.x * 256 + tid;
    int dof  = (job >> 4) & 7;
    float g  = goal[dof];
    const float dt = 0.01f, ax = 25.0f, bx = 6.25f;
    float x = sx[job], dx = sdx[job];

    uint32_t ldsBase   = (uint32_t)(uintptr_t)(&lds[0]);
    uint32_t blockByte = (uint32_t)(blockIdx.x * 256 * TC) * 4u;

    for (int w = 0; w < TC / 32; ++w) {
#if USE_ASYNC_LDS
#pragma unroll 8
        for (int i = 0; i < 32; ++i) {
            int jl = wv * 32 + i;
            async_g2l_b32(ldsBase + (uint32_t)(jl * 33 + lane) * 4u,
                          blockByte + (uint32_t)(jl * TC + w * 32 + lane) * 4u,
                          torque);
        }
        wait_async0();
#else
#pragma unroll 4
        for (int i = 0; i < 32; ++i) {
            int jl = wv * 32 + i;
            lds[jl * 33 + lane] =
                torque[(size_t)blockIdx.x * 256 * TC + (size_t)jl * TC + w * 32 + lane];
        }
#endif
        __syncthreads();
        float* fr = &lds[tid * 33];
#pragma unroll
        for (int tt = 0; tt < 32; ++tt) {
            float ddx = ax * (bx * (g - x) - dx) + fr[tt];
            dx += ddx * dt;
            x  += dx * dt;
            fr[tt] = x;                                    // overwrite in place
        }
        __syncthreads();
#if USE_ASYNC_LDS
#pragma unroll 8
        for (int i = 0; i < 32; ++i) {
            int jl = wv * 32 + i;
            async_l2g_b32(blockByte + (uint32_t)(jl * TC + w * 32 + lane) * 4u,
                          ldsBase + (uint32_t)(jl * 33 + lane) * 4u,
                          out);
        }
        wait_async0();                                     // LDS safe to reuse
#else
#pragma unroll 4
        for (int i = 0; i < 32; ++i) {
            int jl = wv * 32 + i;
            out[(size_t)blockIdx.x * 256 * TC + (size_t)jl * TC + w * 32 + lane] =
                lds[jl * 33 + lane];
        }
#endif
        __syncthreads();
    }
}

// ---------------------------------------------------------------------------
// Launch
// ---------------------------------------------------------------------------
extern "C" void kernel_launch(void* const* d_in, const int* in_sizes, int n_in,
                              void* d_out, int out_size, void* d_ws, size_t ws_size,
                              hipStream_t stream)
{
    const int*   cls  = (const int*)  d_in[0];
    const float* x0   = (const float*)d_in[1];
    const float* goal = (const float*)d_in[2];
    const float* z    = (const float*)d_in[3];
    const float* emb  = (const float*)d_in[4];
    const float* W1   = (const float*)d_in[5];
    const float* b1   = (const float*)d_in[6];
    const float* W2   = (const float*)d_in[7];
    const float* b2   = (const float*)d_in[8];
    const float* W3   = (const float*)d_in[9];
    const float* b3   = (const float*)d_in[10];
    const float* mn   = (const float*)d_in[11];
    const float* mx   = (const float*)d_in[12];
    float* out = (float*)d_out;

    // Workspace carve-up (all 256B aligned)
    uint8_t* ws = (uint8_t*)d_ws;
    float*    h1  = (float*)   (ws + 0);          // 512*64*4   = 131072
    _Float16* h2h = (_Float16*)(ws + 131072);     // 512*128*2  = 131072
    _Float16* W3p = (_Float16*)(ws + 262144);     // 128*32768*2= 8388608
    float*    tq  = (float*)   (ws + 8650752);    // 512*32768*4= 67108864
    float*    qx  = (float*)   (ws + 75759616);   // 65536*4
    float*    qdx = (float*)   (ws + 76021760);
    float*    sx  = (float*)   (ws + 76283904);
    float*    sdx = (float*)   (ws + 76546048);

    // Host: M^TC in double (constant spring-damper transition matrix)
    const double dt = 0.01, axd = 25.0, bxd = 6.25;
    double a00 = 1.0 - axd * bxd * dt * dt;
    double a01 = dt * (1.0 - axd * dt);
    double a10 = -axd * bxd * dt;
    double a11 = 1.0 - axd * dt;
    double p00 = 1, p01 = 0, p10 = 0, p11 = 1;
    for (int i = 0; i < TC; ++i) {
        double n00 = a00 * p00 + a01 * p10;
        double n01 = a00 * p01 + a01 * p11;
        double n10 = a10 * p00 + a11 * p10;
        double n11 = a10 * p01 + a11 * p11;
        p00 = n00; p01 = n01; p10 = n10; p11 = n11;
    }

    k_mlp1  <<<BATCH * H1 / 256, 256, 0, stream>>>(cls, z, emb, W1, b1, h1);
    k_mlp2  <<<BATCH * H2 / 256, 256, 0, stream>>>(h1, W2, b2, h2h);
    k_packW3<<<(NOUT / 16) * 4 * 32 / 256, 256, 0, stream>>>(W3, W3p);
    k_gemm3 <<<(BATCH / 16) * (NOUT / 64) * 32 / 256, 256, 0, stream>>>(
        h2h, W3p, b3, mn, mx, tq);
    k_scan1 <<<NROWS * NCH / 256, 256, 0, stream>>>(tq, goal, qx, qdx);
    k_scan2 <<<NROWS / 256, 256, 0, stream>>>(qx, qdx, x0, sx, sdx,
        (float)p00, (float)p01, (float)p10, (float)p11);
    k_scan3 <<<NROWS * NCH / 256, 256, 0, stream>>>(tq, goal, sx, sdx, out);
}
